// BiLSTM_41455024341762
// MI455X (gfx1250) — matrix operand order repaired and checked by reading, not compiled, not run on hardware
//
#include <hip/hip_runtime.h>
#include <hip/hip_bf16.h>
#include <math.h>

typedef __attribute__((ext_vector_type(16))) _Float16 v16h;
typedef __attribute__((ext_vector_type(8)))  float    v8f;

#define T_LEN 500
#define B_SZ  128
#define H_SZ  256
#define NC    6
#define EMB   200
#define K0PAD 224   // 203 -> padded to 7*32
#define NROWS (T_LEN * B_SZ)   // 64000

__device__ __forceinline__ float sigm(float x) { return 1.0f / (1.0f + __expf(-x)); }

// Async global->LDS 16-byte copy (gfx1250 VGLOBAL GLOBAL_LOAD_ASYNC_TO_LDS_B128,
// tracked by ASYNCcnt). vdst = LDS byte offset, vaddr = 64-bit global address.
__device__ __forceinline__ void async_copy_b128(void* lds_dst, const void* gsrc) {
    unsigned lds_off = (unsigned)(uintptr_t)lds_dst;             // addr[31:0] = LDS offset
    unsigned long long ga = (unsigned long long)(uintptr_t)gsrc;
    asm volatile("global_load_async_to_lds_b128 %0, %1, off"
                 :: "v"(lds_off), "v"(ga) : "memory");
}
__device__ __forceinline__ void async_wait() {
    asm volatile("s_wait_asynccnt 0x0" ::: "memory");
}

// ---------------------------------------------------------------------------
// Embedding + capitalization concat -> time-major f16 x [T, B, 224]
// ---------------------------------------------------------------------------
__global__ void embed_kernel(const int* __restrict__ words, const int* __restrict__ caps,
                             const float* __restrict__ emb, const float* __restrict__ cap_table,
                             _Float16* __restrict__ x) {
    int idx = blockIdx.x * blockDim.x + threadIdx.x;
    if (idx >= NROWS * K0PAD) return;
    int k  = idx % K0PAD;
    int tb = idx / K0PAD;
    int b  = tb % B_SZ;
    int t  = tb / B_SZ;
    float v = 0.0f;
    if (k < EMB) {
        v = emb[(size_t)words[b * T_LEN + t] * EMB + k];
    } else if (k < EMB + 3) {
        v = cap_table[caps[b * T_LEN + t] * 3 + (k - EMB)];
    }
    x[idx] = (_Float16)v;
}

// ---------------------------------------------------------------------------
// Weight conversion f32 -> f16, with row window + zero padding
// dst is [rows_out, 1024]
// ---------------------------------------------------------------------------
__global__ void convert_w(const float* __restrict__ src, _Float16* __restrict__ dst,
                          int src_off, int rows_copy, int rows_out) {
    int idx = blockIdx.x * blockDim.x + threadIdx.x;
    if (idx >= rows_out * 1024) return;
    int r = idx >> 10;
    int c = idx & 1023;
    float v = (r < rows_copy) ? src[(size_t)(src_off + r) * 1024 + c] : 0.0f;
    dst[idx] = (_Float16)v;
}

__global__ void fill_f32(float* p, int n, float v) {
    int i = blockIdx.x * blockDim.x + threadIdx.x;
    if (i < n) p[i] = v;
}
__global__ void fill_f16(_Float16* p, int n, float v) {
    int i = blockIdx.x * blockDim.x + threadIdx.x;
    if (i < n) p[i] = (_Float16)v;
}

// ---------------------------------------------------------------------------
// Big input-projection GEMM: C[M,1024] = A[M,K] (f16) x W[K,1024] (f16) + bias
// Workgroup: 256 threads = 8 waves, tile 64(M) x 128(N).
// Waves: 4 row-waves x 2 col-waves; each wave -> 16x64 (4 WMMA accumulators).
// Tiles staged via async global->LDS (ASYNCcnt), freeing VGPRs for the MMA loop.
// ---------------------------------------------------------------------------
__global__ void wmma_gemm_bias(const _Float16* __restrict__ A, const _Float16* __restrict__ W,
                               const float* __restrict__ bias, float* __restrict__ C,
                               int K) {
    __shared__ __align__(16) _Float16 As[64 * 32];
    __shared__ __align__(16) _Float16 Bs[32 * 128];

    const int tid    = threadIdx.x;
    const int lane   = tid & 31;
    const int waveId = tid >> 5;
    const int wr     = waveId >> 1;   // 0..3
    const int wc     = waveId & 1;    // 0..1
    const int hl     = lane & 15;
    const int hi     = lane >> 4;
    const int rowBase = blockIdx.x * 64;
    const int colBase = blockIdx.y * 128;

    v8f acc[4] = {};

    for (int k0 = 0; k0 < K; k0 += 32) {
        // async load A tile 64x32 (one 16B chunk / thread)
        {
            int r  = tid >> 2;
            int kg = (tid & 3) * 8;
            const _Float16* src = A + (size_t)(rowBase + r) * K + k0 + kg;
            async_copy_b128(&As[r * 32 + kg], src);
            __builtin_prefetch(src + 32, 0, 1);   // global_prefetch next K chunk
        }
        // async load B tile 32x128 (two 16B chunks / thread)
        {
            int r  = tid >> 3;
            int cg = (tid & 7) * 16;
            const _Float16* src = W + (size_t)(k0 + r) * 1024 + colBase + cg;
            async_copy_b128(&Bs[r * 128 + cg],     src);
            async_copy_b128(&Bs[r * 128 + cg + 8], src + 8);
        }
        async_wait();
        __syncthreads();

        // A fragment (16x32): lane<16 row=hl K={0..7,16..23}; lane>=16 K={8..15,24..31}
        v16h af;
        {
            int r = wr * 16 + hl;
            #pragma unroll
            for (int p = 0; p < 8; ++p) {
                int kk = ((p < 4) ? (2 * p) : (16 + 2 * (p - 4))) + 8 * hi;
                af[2 * p]     = As[r * 32 + kk];
                af[2 * p + 1] = As[r * 32 + kk + 1];
            }
        }
        #pragma unroll
        for (int nt = 0; nt < 4; ++nt) {
            // B fragment (32x16): col N=hl; lane<16 K={0..7,16..23}; lane>=16 K={8..15,24..31}
            v16h bf;
            int cc = wc * 64 + nt * 16;
            #pragma unroll
            for (int j = 0; j < 16; ++j) bf[j] = Bs[((j & 7) + ((j >> 3) << 4) + 8 * hi) * 128 + cc + hl];
            acc[nt] = __builtin_amdgcn_wmma_f32_16x16x32_f16(
                false, af, false, bf, (short)0, acc[nt], false, false);
        }
        __syncthreads();
    }

    // C layout: M = v + 8*hi, N = hl
    #pragma unroll
    for (int nt = 0; nt < 4; ++nt) {
        #pragma unroll
        for (int v = 0; v < 8; ++v) {
            int r = rowBase + wr * 16 + v + 8 * hi;
            int c = colBase + wc * 64 + nt * 16 + hl;
            C[(size_t)r * 1024 + c] = acc[nt][v] + bias[c];
        }
    }
}

// ---------------------------------------------------------------------------
// One LSTM timestep:  z = pre[t] + h_prev @ Wh ; gates ; c,h update.
// Grid: 8 workgroups x 256 thr. Each WG owns a 32-unit strip of ALL 4 gates
// (cols n0..n0+31 of i/j/f/o) for all 128 batch rows, so the gate math stays
// in registers. 8 waves: wave w -> batch rows w*16..w*16+15, 8 accumulators.
// Optionally folds the time-max-pool (hmax != nullptr, [B,512] with dir offset).
// ---------------------------------------------------------------------------
__global__ void lstm_step(const float* __restrict__ pre_t,     // [B,1024]
                          const _Float16* __restrict__ Wh,      // [256,1024] f16
                          const _Float16* __restrict__ h_prev,  // [B,256]   f16
                          float* __restrict__ c_state,          // [B,256]   f32
                          _Float16* __restrict__ h_out,         // [B,256]   f16
                          float* __restrict__ hmax) {
    __shared__ __align__(16) _Float16 As[128 * 32];
    __shared__ __align__(16) _Float16 Bs[32 * 128];

    const int tid    = threadIdx.x;
    const int lane   = tid & 31;
    const int waveId = tid >> 5;
    const int hl     = lane & 15;
    const int hi     = lane >> 4;
    const int n0     = blockIdx.x * 32;   // hidden-unit strip

    v8f acc[8];
    // init accumulators from precomputed input projection (includes bias)
    #pragma unroll
    for (int g = 0; g < 4; ++g)
        #pragma unroll
        for (int s = 0; s < 2; ++s) {
            v8f t;
            #pragma unroll
            for (int v = 0; v < 8; ++v) {
                int r = waveId * 16 + v + 8 * hi;
                int c = g * 256 + n0 + s * 16 + hl;
                t[v] = pre_t[(size_t)r * 1024 + c];
            }
            acc[g * 2 + s] = t;
        }

    for (int k0 = 0; k0 < 256; k0 += 32) {
        // async load h_prev tile 128x32
        {
            int r  = tid >> 1;
            int kg = (tid & 1) * 16;
            const _Float16* src = h_prev + (size_t)r * 256 + k0 + kg;
            async_copy_b128(&As[r * 32 + kg],     src);
            async_copy_b128(&As[r * 32 + kg + 8], src + 8);
        }
        // async load Wh tile 32 x (4 gates x 32 cols)
        {
            int r      = tid >> 3;
            int cg     = (tid & 7) * 16;
            int gate   = cg >> 5;
            int within = cg & 31;
            const _Float16* src = Wh + (size_t)(k0 + r) * 1024 + gate * 256 + n0 + within;
            async_copy_b128(&Bs[r * 128 + cg],     src);
            async_copy_b128(&Bs[r * 128 + cg + 8], src + 8);
        }
        async_wait();
        __syncthreads();

        v16h af;
        {
            int r = waveId * 16 + hl;
            #pragma unroll
            for (int p = 0; p < 8; ++p) {
                int kk = ((p < 4) ? (2 * p) : (16 + 2 * (p - 4))) + 8 * hi;
                af[2 * p]     = As[r * 32 + kk];
                af[2 * p + 1] = As[r * 32 + kk + 1];
            }
        }
        #pragma unroll
        for (int t8 = 0; t8 < 8; ++t8) {
            v16h bf;
            #pragma unroll
            for (int j = 0; j < 16; ++j) bf[j] = Bs[((j & 7) + ((j >> 3) << 4) + 8 * hi) * 128 + t8 * 16 + hl];
            acc[t8] = __builtin_amdgcn_wmma_f32_16x16x32_f16(
                false, af, false, bf, (short)0, acc[t8], false, false);
        }
        __syncthreads();
    }

    // gate math + cell update (each (b,u) element owned by exactly one lane)
    #pragma unroll
    for (int s = 0; s < 2; ++s) {
        #pragma unroll
        for (int v = 0; v < 8; ++v) {
            int b = waveId * 16 + v + 8 * hi;
            int u = n0 + s * 16 + hl;
            float zi = acc[0 * 2 + s][v];
            float zj = acc[1 * 2 + s][v];
            float zf = acc[2 * 2 + s][v];
            float zo = acc[3 * 2 + s][v];
            float c_old = c_state[b * 256 + u];
            float c_new = sigm(zf + 1.0f) * c_old + sigm(zi) * tanhf(zj);
            float h     = sigm(zo) * tanhf(c_new);
            c_state[b * 256 + u] = c_new;
            h_out[b * 256 + u]   = (_Float16)h;
            if (hmax) {
                float* p = &hmax[b * 512 + u];
                *p = fmaxf(*p, h);
            }
        }
    }
}

// ---------------------------------------------------------------------------
// Dense head (tiny: 128x64 K=512 and 128x6 K=64; plain VALU is fine)
// ---------------------------------------------------------------------------
__global__ void dense1_elu(const float* __restrict__ hmax, const float* __restrict__ W,
                           const float* __restrict__ b, float* __restrict__ h1) {
    int idx = blockIdx.x * blockDim.x + threadIdx.x;
    if (idx >= 128 * 64) return;
    int bb = idx / 64, j = idx % 64;
    float s = b[j];
    for (int k = 0; k < 512; ++k) s += hmax[bb * 512 + k] * W[k * 64 + j];
    h1[idx] = (s > 0.0f) ? s : (__expf(s) - 1.0f);
}

__global__ void dense2_sig(const float* __restrict__ h1, const float* __restrict__ W,
                           const float* __restrict__ b, float* __restrict__ out) {
    int idx = blockIdx.x * blockDim.x + threadIdx.x;
    if (idx >= 128 * NC) return;
    int bb = idx / NC, j = idx % NC;
    float s = b[j];
    for (int k = 0; k < 64; ++k) s += h1[bb * 64 + k] * W[k * NC + j];
    out[idx] = sigm(s);
}

// ---------------------------------------------------------------------------
// Host orchestration
// ---------------------------------------------------------------------------
static inline size_t alignUp(size_t x) { return (x + 255) & ~(size_t)255; }

extern "C" void kernel_launch(void* const* d_in, const int* in_sizes, int n_in,
                              void* d_out, int out_size, void* d_ws, size_t ws_size,
                              hipStream_t stream) {
    const int*   words     = (const int*)d_in[0];
    const int*   capitals  = (const int*)d_in[1];
    const float* emb       = (const float*)d_in[2];
    const float* cap_table = (const float*)d_in[3];
    // W/b pairs: idx 4..15  (Wf0,bf0,Wf1,bf1,Wf2,bf2,Wb0,bb0,Wb1,bb1,Wb2,bb2)
    const float* d1_W = (const float*)d_in[16];
    const float* d1_b = (const float*)d_in[17];
    const float* d2_W = (const float*)d_in[18];
    const float* d2_b = (const float*)d_in[19];
    float* out = (float*)d_out;

    // ---- workspace carve ----
    char* p = (char*)d_ws;
    size_t off = 0;
    auto carve = [&](size_t bytes) { void* q = p + off; off = alignUp(off + bytes); return q; };

    _Float16* x_f16 = (_Float16*)carve((size_t)NROWS * K0PAD * 2);
    _Float16* hseqA = (_Float16*)carve((size_t)NROWS * H_SZ * 2);
    _Float16* hseqB = (_Float16*)carve((size_t)NROWS * H_SZ * 2);
    float*    pre   = (float*)   carve((size_t)NROWS * 1024 * 4);
    _Float16* WxF[6]; _Float16* WhF[6];
    for (int w = 0; w < 6; ++w) WxF[w] = (_Float16*)carve((size_t)256 * 1024 * 2);
    for (int w = 0; w < 6; ++w) WhF[w] = (_Float16*)carve((size_t)256 * 1024 * 2);
    float*    c_state = (float*)   carve((size_t)B_SZ * H_SZ * 4);
    _Float16* zh      = (_Float16*)carve((size_t)B_SZ * H_SZ * 2);
    float*    hmax    = (float*)   carve((size_t)B_SZ * 2 * H_SZ * 4);
    float*    h1      = (float*)   carve((size_t)B_SZ * 64 * 4);
    (void)ws_size;

    // ---- precompute: embedding, weight conversion, state init ----
    {
        int n = NROWS * K0PAD;
        embed_kernel<<<(n + 255) / 256, 256, 0, stream>>>(words, capitals, emb, cap_table, x_f16);
    }
    for (int w = 0; w < 6; ++w) {
        int l = w % 3;
        const float* Wsrc = (const float*)d_in[4 + 2 * w];
        int kp = (l == 0) ? K0PAD : 256;
        int xcopy = (l == 0) ? 203 : 256;
        int hoff  = (l == 0) ? 203 : 256;
        int n1 = kp * 1024;
        convert_w<<<(n1 + 255) / 256, 256, 0, stream>>>(Wsrc, WxF[w], 0, xcopy, kp);
        int n2 = 256 * 1024;
        convert_w<<<(n2 + 255) / 256, 256, 0, stream>>>(Wsrc, WhF[w], hoff, 256, 256);
    }
    fill_f16<<<(B_SZ * H_SZ + 255) / 256, 256, 0, stream>>>(zh, B_SZ * H_SZ, 0.0f);
    fill_f32<<<(B_SZ * 2 * H_SZ + 255) / 256, 256, 0, stream>>>(hmax, B_SZ * 2 * H_SZ, -2.0f);

    // ---- bidirectional 3-layer LSTM ----
    for (int dir = 0; dir < 2; ++dir) {
        const _Float16* seq_in = x_f16;
        int Kp = K0PAD;
        for (int l = 0; l < 3; ++l) {
            int w = dir * 3 + l;
            const float* bias = (const float*)d_in[5 + 2 * w];
            _Float16* seq_out = (l & 1) ? hseqB : hseqA;

            // input projection for all T at once: pre = seq_in @ Wx + b
            dim3 grid(NROWS / 64, 1024 / 128);
            wmma_gemm_bias<<<grid, 256, 0, stream>>>(seq_in, WxF[w], bias, pre, Kp);

            fill_f32<<<(B_SZ * H_SZ + 255) / 256, 256, 0, stream>>>(c_state, B_SZ * H_SZ, 0.0f);

            float* hmax_ptr = (l == 2) ? (hmax + dir * H_SZ) : nullptr;
            for (int s = 0; s < T_LEN; ++s) {
                // backward direction reverses time only at layer 0 (via pre index);
                // deeper layers consume the already-reversed sequence; max-pool is
                // order-invariant so no final re-reversal is needed.
                int tp = (dir == 1 && l == 0) ? (T_LEN - 1 - s) : s;
                const _Float16* hp = (s == 0) ? zh : (seq_out + (size_t)(s - 1) * B_SZ * H_SZ);
                lstm_step<<<8, 256, 0, stream>>>(pre + (size_t)tp * B_SZ * 1024, WhF[w], hp,
                                                 c_state, seq_out + (size_t)s * B_SZ * H_SZ,
                                                 hmax_ptr);
            }
            seq_in = seq_out;
            Kp = 256;
        }
    }

    // ---- dense head ----
    dense1_elu<<<(128 * 64 + 255) / 256, 256, 0, stream>>>(hmax, d1_W, d1_b, h1);
    dense2_sig<<<(128 * NC + 255) / 256, 256, 0, stream>>>(h1, d2_W, d2_b, out);
}
